// DVEstimator_11914239279929
// MI455X (gfx1250) — matrix operand order repaired and checked by reading, not compiled, not run on hardware
//
#include <hip/hip_runtime.h>
#include <hip/hip_bf16.h>

typedef __attribute__((ext_vector_type(16))) _Float16 v16h;
typedef __attribute__((ext_vector_type(8)))  _Float16 v8h;
typedef __attribute__((ext_vector_type(4)))  _Float16 v4h;
typedef __attribute__((ext_vector_type(8)))  float    v8f;

// 16x32 f16 A-fragment from a row-major LDS row (16B-aligned):
// lane L: row = L&15, akb = (L>=16)*8; a[0..7] = k akb..akb+7,
// a[8..15] = k 16+akb..16+akb+7  -> two ds_load_b128.
__device__ __forceinline__ v16h load_a_frag(const _Float16* arow, int kb, int akb) {
  const v8h lo = *(const v8h*)(arow + kb + akb);
  const v8h hi = *(const v8h*)(arow + kb + akb + 16);
  v16h a;
#pragma unroll
  for (int i = 0; i < 8; ++i) { a[i] = lo[i]; a[8 + i] = hi[i]; }
  return a;
}

// ---------------------------------------------------------------------------
// K0: weight convert: W f32 [Nn,K] -> Wh fp16 [Npad,Kpad] zero-padded.
// ---------------------------------------------------------------------------
__global__ void wcvt_kernel(const float* __restrict__ W, _Float16* __restrict__ Wh,
                            int Nn, int K, int Npad, int Kpad)
{
  const int total = Npad * Kpad;
  for (int idx = blockIdx.x * blockDim.x + threadIdx.x; idx < total;
       idx += gridDim.x * blockDim.x) {
    const int n = idx / Kpad, k = idx - n * Kpad;
    Wh[idx] = (n < Nn && k < K) ? (_Float16)W[n * K + k] : (_Float16)0.f;
  }
}

// ---------------------------------------------------------------------------
// K1: conv1 (1->6, 5x5, pad 2) + bias + relu + 2x2 maxpool, fp16 out. VALU.
// ---------------------------------------------------------------------------
__global__ __launch_bounds__(256) void conv1_kernel(
    const float* __restrict__ X, const float* __restrict__ w,
    const float* __restrict__ b, _Float16* __restrict__ a1)
{
  __shared__ float xs[784];
  __shared__ float ws[150];
  __shared__ float bs[6];
  const int img = blockIdx.x;
  const int tid = threadIdx.x;
  const float* xg = X + (size_t)img * 784;
  for (int i = tid; i < 784; i += 256) xs[i] = xg[i];
  for (int i = tid; i < 150; i += 256) ws[i] = w[i];
  if (tid < 6) bs[tid] = b[tid];
  __syncthreads();
  for (int i = tid; i < 6 * 196; i += 256) {
    const int cch = i / 196, r = i - cch * 196;
    const int py = r / 14, px = r - py * 14;
    const float* wc = &ws[cch * 25];
    const float bias = bs[cch];
    float best = 0.f;  // relu floor
#pragma unroll
    for (int dy = 0; dy < 2; ++dy)
#pragma unroll
      for (int dx = 0; dx < 2; ++dx) {
        const int oy = 2 * py + dy, ox = 2 * px + dx;
        float acc = bias;
#pragma unroll
        for (int ky = 0; ky < 5; ++ky) {
          const int iy = oy + ky - 2;
          if (iy < 0 || iy >= 28) continue;
#pragma unroll
          for (int kx = 0; kx < 5; ++kx) {
            const int ix = ox + kx - 2;
            if (ix < 0 || ix >= 28) continue;
            acc = fmaf(wc[ky * 5 + kx], xs[iy * 28 + ix], acc);
          }
        }
        best = fmaxf(best, acc);
      }
    a1[(size_t)img * 1176 + i] = (_Float16)best;
  }
}

// ---------------------------------------------------------------------------
// K2: conv2 implicit GEMM. M=100 pixels (7 tiles of 16, padded to 112),
// N=16 ch, K=150->160. im2col staged via precomputed index tables; WMMA loop
// is pure ds_load_b128 + global b128 + wmma (5 static WMMAs, unconditional).
// ---------------------------------------------------------------------------
__global__ __launch_bounds__(256) void conv2_wmma_kernel(
    const _Float16* __restrict__ a1, const _Float16* __restrict__ Whc,
    const float* __restrict__ b, _Float16* __restrict__ a2)
{
  __shared__ _Float16 img[1176];        // [6][14][14]
  __shared__ _Float16 Aim[112 * 160];   // im2col: [pixel][k], zero-padded
  __shared__ float convs[112 * 16];     // [pixel][outch]
  __shared__ int idxTab[160];           // k -> ci*196 + ky*14 + kx (or -1)
  __shared__ int pixOff[112];           // p -> oy*14 + ox (or -1)
  const int nimg = blockIdx.x;
  const int tid = threadIdx.x;
  const _Float16* src = a1 + (size_t)nimg * 1176;
  for (int i = tid; i < 1176; i += 256) img[i] = src[i];
  if (tid < 160) {
    int v = -1;
    if (tid < 150) {
      const int ci = tid / 25, rr = tid - ci * 25;
      v = ci * 196 + (rr / 5) * 14 + (rr - (rr / 5) * 5);
    }
    idxTab[tid] = v;
  }
  if (tid < 112)
    pixOff[tid] = (tid < 100) ? (tid / 10) * 14 + (tid - (tid / 10) * 10) : -1;
  __syncthreads();
  for (int i = tid; i < 112 * 160; i += 256) {
    const int p = i / 160, k = i - p * 160;
    const int ia = idxTab[k], pb = pixOff[p];
    Aim[i] = (ia >= 0 && pb >= 0) ? img[ia + pb] : (_Float16)0.f;
  }
  __syncthreads();

  const int wave = tid >> 5;
  const int lane = tid & 31;
  if (wave < 7) {                        // wave-uniform: EXEC all-ones
    const int row = lane & 15;
    const int akb = (lane >> 4) << 3;    // 0 or 8
    const int bkb = (lane >> 4) << 4;    // 0 or 16
    const _Float16* arow = &Aim[(wave * 16 + row) * 160];
    const _Float16* wrow = Whc + row * 160;
    v8f c = {};
#pragma unroll
    for (int ks = 0; ks < 5; ++ks) {
      const int kb = ks * 32;
      const v16h a = load_a_frag(arow, kb, akb);
      const v16h bb = load_a_frag(wrow, kb, akb);
      c = __builtin_amdgcn_wmma_f32_16x16x32_f16(false, a, false, bb,
                                                 (short)0, c, false, false);
    }
    const int pbase = wave * 16 + ((lane >> 4) << 3);
#pragma unroll
    for (int v = 0; v < 8; ++v)          // p <= 111 < 112: always in-bounds
      convs[(pbase + v) * 16 + row] = c[v];
  }
  __syncthreads();
  for (int i = tid; i < 400; i += 256) {   // bias+relu+2x2 maxpool
    const int cch = i / 25, r = i - cch * 25;
    const int py = r / 5, px = r - py * 5;
    const float m0 = convs[((2 * py) * 10 + 2 * px) * 16 + cch];
    const float m1 = convs[((2 * py) * 10 + 2 * px + 1) * 16 + cch];
    const float m2 = convs[((2 * py + 1) * 10 + 2 * px) * 16 + cch];
    const float m3 = convs[((2 * py + 1) * 10 + 2 * px + 1) * 16 + cch];
    const float pv = fmaxf(fmaxf(m0, m1), fmaxf(m2, m3)) + b[cch];
    a2[(size_t)nimg * 400 + i] = (_Float16)fmaxf(pv, 0.f);
  }
}

// ---------------------------------------------------------------------------
// K3..K5: fc GEMM, KSTEPS compile-time so the k-loop fully unrolls.
// A fp16 [M rows, lda] staged to 64-row LDS tile in b64 chunks (K%4==0);
// B fp16 [Npad][KP]; output padded [Mpad][ldo] so stores are unconditional.
// ---------------------------------------------------------------------------
template <int KSTEPS, bool RELU, bool OUTF32>
__global__ __launch_bounds__(256) void wmma_fc_kernel(
    const _Float16* __restrict__ A, const _Float16* __restrict__ Wh,
    const float* __restrict__ bias, void* __restrict__ outp,
    int M, int K, int lda, int Nn, int ldo)
{
  constexpr int KP = KSTEPS * 32;
  constexpr int CPR = KP / 4;        // 4-half chunks per row
  extern __shared__ _Float16 As[];   // [64][KP]
  const int tid = threadIdx.x;
  const int mblock = blockIdx.x * 64;
  for (int idx = tid; idx < 64 * CPR; idx += 256) {
    const int r = idx / CPR, c = idx - r * CPR;
    const int k0 = c * 4;
    const int gr = mblock + r;
    v4h val = {};
    if (gr < M && k0 < K) val = *(const v4h*)(A + (size_t)gr * lda + k0);
    *(v4h*)(&As[r * KP + k0]) = val;
  }
  __syncthreads();

  const int wave = tid >> 5;
  const int lane = tid & 31;
  const int colTiles = (Nn + 15) >> 4;
  const int row = lane & 15;
  const int akb = (lane >> 4) << 3;
  const int bkb = (lane >> 4) << 4;

  for (int t = wave; t < 4 * colTiles; t += 8) {   // wave-uniform
    const int rt = t & 3;
    const int ct = t >> 2;
    const int n = (ct << 4) + row;
    const _Float16* arow = &As[(rt * 16 + row) * KP];
    const _Float16* wrow = Wh + (size_t)n * KP;
    v8f c = {};
#pragma unroll
    for (int ks = 0; ks < KSTEPS; ++ks) {
      const int kb = ks * 32;
      const v16h a = load_a_frag(arow, kb, akb);
      const v16h bb = load_a_frag(wrow, kb, akb);
      c = __builtin_amdgcn_wmma_f32_16x16x32_f16(false, a, false, bb,
                                                 (short)0, c, false, false);
    }
    const float bn = (n < Nn) ? bias[n] : 0.f;
    const int mbase = mblock + rt * 16 + ((lane >> 4) << 3);
#pragma unroll
    for (int v = 0; v < 8; ++v) {        // padded out: always in-bounds
      float val = c[v] + bn;
      if (RELU) val = fmaxf(val, 0.f);
      if (OUTF32) ((float*)outp)[(size_t)(mbase + v) * ldo + n] = val;
      else ((_Float16*)outp)[(size_t)(mbase + v) * ldo + n] = (_Float16)val;
    }
  }
}

// K6: per_point[i] = logits[i, y[i]]  (logits padded to ld 16)
__global__ void gather_kernel(const float* __restrict__ logits,
                              const int* __restrict__ y,
                              float* __restrict__ pp, int N)
{
  const int i = blockIdx.x * blockDim.x + threadIdx.x;
  if (i < N) pp[i] = logits[(size_t)i * 16 + y[i]];
}

// ---------------------------------------------------------------------------
// K7: diff[s] = subsets[s,:N] . per_point - ues[s] + subsets[s,N]*null_value.
// 102 MB byte stream (aligned uint4 + prefetch); per_point hot in L2.
// ---------------------------------------------------------------------------
__global__ __launch_bounds__(256) void subset_kernel(
    const unsigned char* __restrict__ subsets, const float* __restrict__ pp,
    const float* __restrict__ ues, const float* __restrict__ nullv,
    float* __restrict__ diff, int N)
{
  __shared__ float red[256];
  const int s = blockIdx.x;
  const int tid = threadIdx.x;
  const unsigned char* row = subsets + (size_t)s * (size_t)(N + 1);
  float local = 0.f;
  int lead = (int)((16 - ((size_t)row & 15)) & 15);
  if (lead > N) lead = N;
  if (tid < lead) local += (float)row[tid] * pp[tid];
  const int nvec = (N - lead) >> 4;
  const uint4* vp = (const uint4*)(row + lead);
  for (int vi = tid; vi < nvec; vi += 256) {
    if (vi + 256 < nvec) __builtin_prefetch(vp + vi + 256, 0, 1);
    const uint4 wrd = vp[vi];
    const int base = lead + vi * 16;
    const unsigned u[4] = {wrd.x, wrd.y, wrd.z, wrd.w};
#pragma unroll
    for (int q = 0; q < 4; ++q)
#pragma unroll
      for (int j = 0; j < 4; ++j)
        local = fmaf((float)((u[q] >> (8 * j)) & 0xffu), pp[base + q * 4 + j], local);
  }
  for (int j = lead + (nvec << 4) + tid; j < N; j += 256)
    local += (float)row[j] * pp[j];
  red[tid] = local;
  __syncthreads();
  for (int st = 128; st > 0; st >>= 1) {
    if (tid < st) red[tid] += red[tid + st];
    __syncthreads();
  }
  if (tid == 0) diff[s] = red[0] - ues[s] + (float)row[N] * nullv[0];
}

// K8: out = mean(diff^2)
__global__ __launch_bounds__(256) void mse_kernel(const float* __restrict__ diff,
                                                  float* __restrict__ out, int S)
{
  __shared__ float red[256];
  const int tid = threadIdx.x;
  float local = 0.f;
  for (int i = tid; i < S; i += 256) local += diff[i] * diff[i];
  red[tid] = local;
  __syncthreads();
  for (int st = 128; st > 0; st >>= 1) {
    if (tid < st) red[tid] += red[tid + st];
    __syncthreads();
  }
  if (tid == 0) out[0] = red[0] / (float)S;
}

extern "C" void kernel_launch(void* const* d_in, const int* in_sizes, int n_in,
                              void* d_out, int out_size, void* d_ws, size_t ws_size,
                              hipStream_t stream) {
  const unsigned char* subsets = (const unsigned char*)d_in[0];
  const float* ues   = (const float*)d_in[1];
  const float* X     = (const float*)d_in[2];
  const int*   y     = (const int*)d_in[3];
  const float* c1w   = (const float*)d_in[4];
  const float* c1b   = (const float*)d_in[5];
  const float* c2w   = (const float*)d_in[6];
  const float* c2b   = (const float*)d_in[7];
  const float* f1w   = (const float*)d_in[8];
  const float* f1b   = (const float*)d_in[9];
  const float* f2w   = (const float*)d_in[10];
  const float* f2b   = (const float*)d_in[11];
  const float* f3w   = (const float*)d_in[12];
  const float* f3b   = (const float*)d_in[13];
  const float* nullv = (const float*)d_in[14];

  const int N = in_sizes[3];  // 50000 images
  const int S = in_sizes[1];  // 2048 subsets
  const int mb = (N + 63) / 64;
  const int Mpad = mb * 64;

  char* ws = (char*)d_ws;
  size_t off = 0;
  auto carve = [&](size_t bytes) {
    void* p = ws + off;
    off = (off + bytes + 255) & ~(size_t)255;
    return p;
  };
  _Float16* a1     = (_Float16*)carve((size_t)N * 1176 * sizeof(_Float16));
  _Float16* a2     = (_Float16*)carve((size_t)N * 400  * sizeof(_Float16));
  _Float16* a3     = (_Float16*)carve((size_t)Mpad * 128 * sizeof(_Float16));
  _Float16* a4     = (_Float16*)carve((size_t)Mpad * 96  * sizeof(_Float16));
  float*    logits = (float*)   carve((size_t)Mpad * 16  * sizeof(float));
  float*    pp     = (float*)   carve((size_t)N          * sizeof(float));
  float*    diff   = (float*)   carve((size_t)S          * sizeof(float));
  _Float16* Whc2   = (_Float16*)carve((size_t)16  * 160 * sizeof(_Float16));
  _Float16* Wh1    = (_Float16*)carve((size_t)128 * 416 * sizeof(_Float16));
  _Float16* Wh2    = (_Float16*)carve((size_t)96  * 128 * sizeof(_Float16));
  _Float16* Wh3    = (_Float16*)carve((size_t)16  * 96  * sizeof(_Float16));
  (void)ws_size; (void)n_in; (void)out_size;

  // Pre-convert all weights to zero-padded fp16 (tiny, L2-resident).
  wcvt_kernel<<<16, 256, 0, stream>>>(c2w, Whc2, 16, 150, 16, 160);
  wcvt_kernel<<<64, 256, 0, stream>>>(f1w, Wh1, 120, 400, 128, 416);
  wcvt_kernel<<<16, 256, 0, stream>>>(f2w, Wh2, 84, 120, 96, 128);
  wcvt_kernel<<<4, 256, 0, stream>>>(f3w, Wh3, 10, 84, 16, 96);

  conv1_kernel<<<N, 256, 0, stream>>>(X, c1w, c1b, a1);
  conv2_wmma_kernel<<<N, 256, 0, stream>>>(a1, Whc2, c2b, a2);

  wmma_fc_kernel<13, true, false><<<mb, 256, (size_t)64 * 416 * sizeof(_Float16), stream>>>(
      a2, Wh1, f1b, a3, N, 400, 400, 120, 128);
  wmma_fc_kernel<4, true, false><<<mb, 256, (size_t)64 * 128 * sizeof(_Float16), stream>>>(
      a3, Wh2, f2b, a4, Mpad, 120, 128, 84, 96);
  wmma_fc_kernel<3, false, true><<<mb, 256, (size_t)64 * 96 * sizeof(_Float16), stream>>>(
      a4, Wh3, f3b, logits, Mpad, 84, 96, 10, 16);

  gather_kernel<<<(N + 255) / 256, 256, 0, stream>>>(logits, y, pp, N);
  subset_kernel<<<S, 256, 0, stream>>>(subsets, pp, ues, nullv, diff, N);
  mse_kernel<<<1, 256, 0, stream>>>(diff, (float*)d_out, S);
}
